// ModulatedChunks_61040075211207
// MI455X (gfx1250) — matrix operand order repaired and checked by reading, not compiled, not run on hardware
//
#include <hip/hip_runtime.h>
#include <hip/hip_bf16.h>
#include <math.h>

// ---------------------------------------------------------------------------
// Problem constants (from reference setup_inputs)
// ---------------------------------------------------------------------------
constexpr int B_  = 4;
constexpr int T_  = 512;
constexpr int C_  = 512;   // all GEMMs here have K = N = 512
constexpr int WN  = 32;
constexpr int Q_  = 512;
constexpr int D_  = 512;
constexpr int P_  = 64;
constexpr int NC  = 8;

typedef __attribute__((ext_vector_type(16))) __bf16 bf16x16;
typedef __attribute__((ext_vector_type(8)))  float  f32x8;
typedef __attribute__((ext_vector_type(2)))  float  f32x2;
typedef int v4i_ __attribute__((vector_size(16)));   // matches builtin param type

union FragBF { unsigned u[8]; bf16x16 v; };

static __device__ __forceinline__ unsigned bf16_rne(float f) {
    unsigned u = __float_as_uint(f);
    return (u + 0x7FFFu + ((u >> 16) & 1u)) >> 16;   // round-to-nearest-even
}
static __device__ __forceinline__ unsigned pack2(float lo, float hi) {
    return bf16_rne(lo) | (bf16_rne(hi) << 16);
}

// CDNA5 async global->LDS path (guarded; falls back to load + ds_store)
#define ASG __attribute__((address_space(1)))
#define ASL __attribute__((address_space(3)))
#if defined(__has_builtin)
#if __has_builtin(__builtin_amdgcn_global_load_async_to_lds_b128)
#define USE_ASYNC_LDS 1
#endif
#if __has_builtin(__builtin_amdgcn_s_wait_asynccnt)
#define ASYNC_WAIT() __builtin_amdgcn_s_wait_asynccnt(0)
#else
#define ASYNC_WAIT() asm volatile("s_wait_asynccnt 0" ::: "memory")
#endif
#endif

// ---------------------------------------------------------------------------
// Shared WMMA inner loop: one wave's 16x16 tile, A panel in LDS, B in global.
//   la: lane's A base in LDS  (= lds_a + ln*256 + h*4), bf16 pairs along K
//   bb: lane's B base         (= Bp + h*4*512 + n), layout [K/2][512]
// K = 512 fully unrolled: all loads become immediate-offset ds/global loads.
// ---------------------------------------------------------------------------
static __device__ __forceinline__ f32x8 wmma_panel(const unsigned* la,
                                                   const unsigned* __restrict__ bb) {
    f32x8 acc = {};
#pragma unroll
    for (int k0 = 0; k0 < 512; k0 += 32) {
        const int k2 = k0 >> 1;
        uint4 a0 = *(const uint4*)(la + k2);        // K = h*8 + 0..7
        uint4 a1 = *(const uint4*)(la + k2 + 8);    // K = 16 + h*8 + 0..7
        FragBF fa, fb;
        fa.u[0] = a0.x; fa.u[1] = a0.y; fa.u[2] = a0.z; fa.u[3] = a0.w;
        fa.u[4] = a1.x; fa.u[5] = a1.y; fa.u[6] = a1.z; fa.u[7] = a1.w;
#pragma unroll
        for (int j = 0; j < 8; ++j) fb.u[j] = bb[(k2 + j + (j >= 4 ? 4 : 0)) * 512];
        acc = __builtin_amdgcn_wmma_f32_16x16x32_bf16(false, fa.v, false, fb.v,
                                                      (short)0, acc, false, false);
    }
    return acc;
}

// ---------------------------------------------------------------------------
// K0: add sinusoidal PE and pack rows to bf16 pairs (pairs along feature dim)
// ---------------------------------------------------------------------------
__global__ void pe_pack_kernel(const float* __restrict__ src, unsigned* __restrict__ dst,
                               int rowsTotal, int Tdim, int Ccols) {
    int idx  = blockIdx.x * blockDim.x + threadIdx.x;
    int half = Ccols >> 1;
    if (idx >= rowsTotal * half) return;
    int i   = idx % half;
    int row = idx / half;
    int t   = row % Tdim;
    float div = expf((float)(2 * i) * (-logf(10000.0f) / (float)Ccols));
    float ang = (float)t * div;
    float lo  = src[row * Ccols + 2 * i]     + sinf(ang);
    float hi  = src[row * Ccols + 2 * i + 1] + cosf(ang);
    dst[idx] = pack2(lo, hi);
}

// K1: pack a row-major f32 weight [512][512] to bf16 pairs along K
__global__ void pack_weight_kernel(const float* __restrict__ w, unsigned* __restrict__ dst) {
    int idx = blockIdx.x * blockDim.x + threadIdx.x;
    if (idx >= 256 * 512) return;
    int n  = idx & 511;
    int k2 = idx >> 9;
    dst[idx] = pack2(w[(2 * k2) * 512 + n], w[(2 * k2 + 1) * 512 + n]);
}

// K2: pack contiguous f32 rows to bf16 pairs along the (even) last dim
__global__ void pack_rows_kernel(const float* __restrict__ src, unsigned* __restrict__ dst,
                                 int totalPairs) {
    int idx = blockIdx.x * blockDim.x + threadIdx.x;
    if (idx >= totalPairs) return;
    dst[idx] = pack2(src[2 * idx], src[2 * idx + 1]);
}

// ---------------------------------------------------------------------------
// K3: bf16 GEMM (N = K = 512), fp32 accumulate, A panel staged via async->LDS.
// Block = 8 waves sharing one mt (16 A rows), covering 8 consecutive nt.
// grid.x = mT * 4   (512/16 nt / 8 waves = 4 groups)
// ---------------------------------------------------------------------------
__global__ void gemm_bf16_kernel(const unsigned* __restrict__ Ap, const unsigned* __restrict__ Bp,
                                 const float* __restrict__ bias, float* __restrict__ Out,
                                 int M, int relu) {
    __shared__ unsigned lds_a[16 * 256];             // 16 rows x 512 bf16 = 16KB
    int mt = blockIdx.x >> 2;
    int group = blockIdx.x & 3;
    {   // cooperative stage of A panel
        int row = threadIdx.x >> 4, part = threadIdx.x & 15;
        int mA = mt * 16 + row; if (mA >= M) mA = M - 1;
        const unsigned* gsrc = Ap + mA * 256 + part * 16;
        unsigned* ldst = &lds_a[row * 256 + part * 16];
#ifdef USE_ASYNC_LDS
#pragma unroll
        for (int j = 0; j < 4; ++j)
            __builtin_amdgcn_global_load_async_to_lds_b128(
                (ASG v4i_*)(gsrc + j * 4), (ASL v4i_*)(ldst + j * 4), 0, 0);
        ASYNC_WAIT();
#else
#pragma unroll
        for (int j = 0; j < 4; ++j)
            *(uint4*)(ldst + j * 4) = *(const uint4*)(gsrc + j * 4);
#endif
    }
    __syncthreads();
    int lane = threadIdx.x & 31, wave = threadIdx.x >> 5;
    int ln = lane & 15, h = lane >> 4;
    int n = (group * 8 + wave) * 16 + ln;
    const unsigned* bb = Bp + h * 4 * 512 + n;
    __builtin_prefetch(bb, 0, 1);
    f32x8 acc = wmma_panel(&lds_a[ln * 256 + h * 4], bb);
    float bv = bias[n];
#pragma unroll
    for (int r = 0; r < 8; ++r) {
        int mm = mt * 16 + h * 8 + r;
        if (mm < M) {
            float x = acc[r] + bv;
            if (relu) x = fmaxf(x, 0.0f);
            Out[mm * 512 + n] = x;
        }
    }
}

// ---------------------------------------------------------------------------
// K4: sim[b][t][w] = sum_c v[b][t][c] * enc1[b][w][c]   (fp32 WMMA 16x16x4)
// ---------------------------------------------------------------------------
__global__ void sim_wmma_kernel(const float* __restrict__ v, const float* __restrict__ enc1,
                                float* __restrict__ sim) {
    int b = blockIdx.z;
    int lane = threadIdx.x & 31, wave = threadIdx.x >> 5;
    const int nT = WN / 16;
    int tile = blockIdx.x * (blockDim.x >> 5) + wave;
    if (tile >= (T_ / 16) * nT) return;
    int mt = tile / nT, nt = tile % nT;
    int ln = lane & 15, h = lane >> 4;
    const float* arow = v    + (b * T_ + mt * 16 + ln) * C_;
    const float* brow = enc1 + (b * WN + nt * 16 + ln) * C_;   // B[k][n]=enc1[n][k]
    f32x8 acc = {};
#if __has_builtin(__builtin_amdgcn_wmma_f32_16x16x4_f32)
#pragma unroll 8
    for (int k0 = 0; k0 < C_; k0 += 4) {
        f32x2 a2 = { arow[k0 + 2 * h], arow[k0 + 2 * h + 1] };
        f32x2 b2 = { brow[k0 + 2 * h], brow[k0 + 2 * h + 1] };
        acc = __builtin_amdgcn_wmma_f32_16x16x4_f32(false, a2, false, b2,
                                                    (short)0, acc, false, false);
    }
#else
    const float* aBase = v + (b * T_ + mt * 16) * C_;
    for (int r = 0; r < 8; ++r) {
        float s = 0.f;
        for (int k = 0; k < C_; ++k) s += aBase[(h * 8 + r) * C_ + k] * brow[k];
        acc[r] = s;
    }
#endif
#pragma unroll
    for (int r = 0; r < 8; ++r)
        sim[(b * T_ + mt * 16 + h * 8 + r) * WN + nt * 16 + ln] = acc[r];
}

// ---------------------------------------------------------------------------
// K5: sliding sums: dst[b][t][w] = sum_{j<cs} src[b][t+j][w], t in [0, T-cs]
// ---------------------------------------------------------------------------
__global__ void sliding_sum_kernel(const float* __restrict__ src, float* __restrict__ dst,
                                   int W, int cs) {
    int idx = blockIdx.x * blockDim.x + threadIdx.x;
    if (idx >= B_ * W) return;
    int b = idx / W, w = idx % W;
    const float* s0 = src + b * T_ * W + w;
    float*       d0 = dst + b * T_ * W + w;
    for (int t = 0; t <= T_ - cs; ++t) {
        float s = 0.f;
        for (int j = 0; j < cs; ++j) s += s0[(t + j) * W];
        d0[t * W] = s;
    }
}

// K6: labels: lab[b][t] = argmax_w slidSim[b][t][w]  (first-max tie-break)
__global__ void argmax_kernel(const float* __restrict__ ss, int* __restrict__ lab, int cs) {
    int idx = blockIdx.x * blockDim.x + threadIdx.x;
    if (idx >= B_ * T_) return;
    int t = idx % T_;
    if (t > T_ - cs) return;
    const float* p = ss + idx * WN;
    float best = p[0]; int bi = 0;
    for (int w = 1; w < WN; ++w) { float x = p[w]; if (x > best) { best = x; bi = w; } }
    lab[idx] = bi;
}

// ---------------------------------------------------------------------------
// K7: proj GEMM with fused A construction staged once per block into LDS:
//   row r = l*NC + k -> t = l + k*cs; a[c] = enc2[b][lab[b][t]][c]*Ssum[b][t][c]/cs
//   Out[b][r][d] = a . proj_w[:,d] + proj_b[d]
// ---------------------------------------------------------------------------
__global__ void proj_gemm_kernel(const float* __restrict__ Ssum, const int* __restrict__ lab,
                                 const float* __restrict__ enc2, const unsigned* __restrict__ Bp,
                                 const float* __restrict__ bias, float* __restrict__ Out,
                                 int rows, int cs) {
    __shared__ unsigned lds_a[16 * 256];
    int b = blockIdx.z;
    int mt = blockIdx.x >> 2;
    int group = blockIdx.x & 3;
    {   // build modulated A panel (bf16 pairs) cooperatively
        int row = threadIdx.x >> 4, part = threadIdx.x & 15;
        int r = mt * 16 + row; if (r >= rows) r = rows - 1;
        int l = r >> 3, k = r & 7;               // NC == 8
        int t = l + k * cs;
        int lb = lab[b * T_ + t];
        const float* e2 = enc2 + (b * WN + lb) * C_ + part * 32;
        const float* ss = Ssum + (b * T_ + t) * C_ + part * 32;
        float inv = 1.0f / (float)cs;
        unsigned* ldst = &lds_a[row * 256 + part * 16];
#pragma unroll
        for (int j = 0; j < 8; ++j) {
            float4 e = *(const float4*)(e2 + j * 4);
            float4 s = *(const float4*)(ss + j * 4);
            ldst[2 * j]     = pack2(e.x * s.x * inv, e.y * s.y * inv);
            ldst[2 * j + 1] = pack2(e.z * s.z * inv, e.w * s.w * inv);
        }
    }
    __syncthreads();
    int lane = threadIdx.x & 31, wave = threadIdx.x >> 5;
    int ln = lane & 15, h = lane >> 4;
    int n = (group * 8 + wave) * 16 + ln;
    f32x8 acc = wmma_panel(&lds_a[ln * 256 + h * 4], Bp + h * 4 * 512 + n);
    float bv = bias[n];
#pragma unroll
    for (int rr = 0; rr < 8; ++rr) {
        int mm = mt * 16 + h * 8 + rr;
        if (mm < rows) Out[(b * rows + mm) * 512 + n] = acc[rr] + bv;
    }
}

// ---------------------------------------------------------------------------
// K8: adaptive max pool: bin p covers rows [floor(p*Lin/P), ceil((p+1)*Lin/P))
// ---------------------------------------------------------------------------
__global__ void pool_kernel(const float* __restrict__ Mbuf, const int* __restrict__ vid_len,
                            float* __restrict__ pooled, int rows) {
    int p = blockIdx.x, b = blockIdx.y;
    int Lin = vid_len[b] * NC; if (Lin < 1) Lin = 1;
    int start = (p * Lin) / P_;
    int end   = ((p + 1) * Lin + P_ - 1) / P_;
    const float* base = Mbuf + b * rows * D_;
    for (int d = threadIdx.x; d < D_; d += blockDim.x) {
        float m = -INFINITY;
        for (int rr = start; rr < end; ++rr) m = fmaxf(m, base[rr * D_ + d]);
        pooled[(b * P_ + p) * D_ + d] = m;
    }
}

// K9: seg head: out[b][i][p] = pooled[b][p][:] . w[:,i] + bias[i]
__global__ void seg_head_kernel(const float* __restrict__ pooled, const float* __restrict__ w,
                                const float* __restrict__ bias, float* __restrict__ out) {
    int idx = blockIdx.x * blockDim.x + threadIdx.x;      // = b*2*P + i*P + p
    if (idx >= B_ * 2 * P_) return;
    int rem = idx % (2 * P_);
    int b = idx / (2 * P_), i = rem / P_, p = rem % P_;
    const float* pr = pooled + (b * P_ + p) * D_;
    float acc = bias[i];
    for (int d = 0; d < D_; ++d) acc += pr[d] * w[d * 2 + i];
    out[idx] = acc;
}

// K10: start/end head: out[b][pp] = sum_{d,p} pooled[b][p][d]*w[d*P+p][pp] + bias[pp]
__global__ void vec_head_kernel(const float* __restrict__ pooled, const float* __restrict__ w,
                                const float* __restrict__ bias, float* __restrict__ out) {
    int pp = blockIdx.x, b = blockIdx.y;
    __shared__ float red[256];
    const float* pb = pooled + b * P_ * D_;
    float acc = 0.f;
    for (int nn = threadIdx.x; nn < D_ * P_; nn += blockDim.x) {
        int p = nn & (P_ - 1), d = nn >> 6;               // nn = d*P + p
        acc += pb[p * D_ + d] * w[nn * P_ + pp];
    }
    red[threadIdx.x] = acc; __syncthreads();
    for (int s = blockDim.x >> 1; s > 0; s >>= 1) {
        if ((int)threadIdx.x < s) red[threadIdx.x] += red[threadIdx.x + s];
        __syncthreads();
    }
    if (threadIdx.x == 0) out[b * P_ + pp] = red[0] + bias[pp];
}

// ---------------------------------------------------------------------------
// Host side
// ---------------------------------------------------------------------------
extern "C" void kernel_launch(void* const* d_in, const int* in_sizes, int n_in,
                              void* d_out, int out_size, void* d_ws, size_t ws_size,
                              hipStream_t stream) {
    (void)in_sizes; (void)n_in; (void)out_size; (void)ws_size;
    const float* vis  = (const float*)d_in[0];
    const float* qry  = (const float*)d_in[1];
    const int*   vlen = (const int*)  d_in[2];
    const float* vw1  = (const float*)d_in[3];  const float* vb1 = (const float*)d_in[4];
    const float* vw2  = (const float*)d_in[5];  const float* vb2 = (const float*)d_in[6];
    const float* s1w1 = (const float*)d_in[7];  const float* s1b1 = (const float*)d_in[8];
    const float* s1w2 = (const float*)d_in[9];  const float* s1b2 = (const float*)d_in[10];
    const float* s2w1 = (const float*)d_in[11]; const float* s2b1 = (const float*)d_in[12];
    const float* s2w2 = (const float*)d_in[13]; const float* s2b2 = (const float*)d_in[14];
    const float* pw   = (const float*)d_in[15]; const float* pb   = (const float*)d_in[16];
    const float* stw0 = (const float*)d_in[17]; const float* stb0 = (const float*)d_in[18];
    const float* enw0 = (const float*)d_in[19]; const float* enb0 = (const float*)d_in[20];
    const float* sgw0 = (const float*)d_in[21]; const float* sgb0 = (const float*)d_in[22];
    const float* stw1 = (const float*)d_in[23]; const float* stb1 = (const float*)d_in[24];
    const float* enw1 = (const float*)d_in[25]; const float* enb1 = (const float*)d_in[26];
    const float* sgw1 = (const float*)d_in[27]; const float* sgb1 = (const float*)d_in[28];
    float* out = (float*)d_out;

    char* ws = (char*)d_ws; size_t off = 0;
    auto alloc = [&](size_t bytes) -> char* {
        char* p = ws + off; off = (off + bytes + 255) & ~(size_t)255; return p;
    };
    const int ROWS0 = (T_ - 64 + 1) * NC;    // 3592
    const int ROWS1 = (T_ - 128 + 1) * NC;   // 3080
    unsigned* XVP  = (unsigned*)alloc((size_t)B_ * T_ * (C_ / 2) * 4);
    unsigned* XQP  = (unsigned*)alloc((size_t)B_ * WN * (Q_ / 2) * 4);
    size_t wpB = (size_t)256 * 512 * 4;
    unsigned* WPv1 = (unsigned*)alloc(wpB); unsigned* WPv2 = (unsigned*)alloc(wpB);
    unsigned* WP11 = (unsigned*)alloc(wpB); unsigned* WP12 = (unsigned*)alloc(wpB);
    unsigned* WP21 = (unsigned*)alloc(wpB); unsigned* WP22 = (unsigned*)alloc(wpB);
    unsigned* WPpj = (unsigned*)alloc(wpB);
    float* H    = (float*)alloc((size_t)B_ * T_ * C_ * 4);
    unsigned* HP = (unsigned*)alloc((size_t)B_ * T_ * (C_ / 2) * 4);
    float* V    = (float*)alloc((size_t)B_ * T_ * C_ * 4);
    float* ENC1 = (float*)alloc((size_t)B_ * WN * C_ * 4);
    float* ENC2 = (float*)alloc((size_t)B_ * WN * C_ * 4);
    float* SIM  = (float*)alloc((size_t)B_ * T_ * WN * 4);
    float* SS8  = (float*)alloc((size_t)B_ * T_ * C_ * 4);
    float* SS16 = (float*)alloc((size_t)B_ * T_ * C_ * 4);
    float* SM8  = (float*)alloc((size_t)B_ * T_ * WN * 4);
    float* SM16 = (float*)alloc((size_t)B_ * T_ * WN * 4);
    int* LAB8   = (int*)alloc((size_t)B_ * T_ * 4);
    int* LAB16  = (int*)alloc((size_t)B_ * T_ * 4);
    float* MBUF = (float*)alloc((size_t)B_ * ROWS0 * D_ * 4);
    float* POOL = (float*)alloc((size_t)B_ * P_ * D_ * 4);

    const int TB = 256;
    auto blk = [](int n, int t) { return (n + t - 1) / t; };
    auto ggrid = [](int M) { return ((M + 15) / 16) * 4; };   // mT * (512/16/8)

    pe_pack_kernel<<<blk(B_ * T_ * (C_ / 2), TB), TB, 0, stream>>>(vis, XVP, B_ * T_, T_, C_);
    pe_pack_kernel<<<blk(B_ * WN * (Q_ / 2), TB), TB, 0, stream>>>(qry, XQP, B_ * WN, WN, Q_);
    int wtot = 256 * 512;
    pack_weight_kernel<<<blk(wtot, TB), TB, 0, stream>>>(vw1,  WPv1);
    pack_weight_kernel<<<blk(wtot, TB), TB, 0, stream>>>(vw2,  WPv2);
    pack_weight_kernel<<<blk(wtot, TB), TB, 0, stream>>>(s1w1, WP11);
    pack_weight_kernel<<<blk(wtot, TB), TB, 0, stream>>>(s1w2, WP12);
    pack_weight_kernel<<<blk(wtot, TB), TB, 0, stream>>>(s2w1, WP21);
    pack_weight_kernel<<<blk(wtot, TB), TB, 0, stream>>>(s2w2, WP22);
    pack_weight_kernel<<<blk(wtot, TB), TB, 0, stream>>>(pw,   WPpj);

    // v = MLP2(vis + PE)
    gemm_bf16_kernel<<<ggrid(B_ * T_), TB, 0, stream>>>(XVP, WPv1, vb1, H, B_ * T_, 1);
    pack_rows_kernel<<<blk(B_ * T_ * (C_ / 2), TB), TB, 0, stream>>>(H, HP, B_ * T_ * (C_ / 2));
    gemm_bf16_kernel<<<ggrid(B_ * T_), TB, 0, stream>>>(HP, WPv2, vb2, V, B_ * T_, 0);
    // enc1 / enc2 = MLP2(q + PE)
    gemm_bf16_kernel<<<ggrid(B_ * WN), TB, 0, stream>>>(XQP, WP11, s1b1, H, B_ * WN, 1);
    pack_rows_kernel<<<blk(B_ * WN * (Q_ / 2), TB), TB, 0, stream>>>(H, HP, B_ * WN * (Q_ / 2));
    gemm_bf16_kernel<<<ggrid(B_ * WN), TB, 0, stream>>>(HP, WP12, s1b2, ENC1, B_ * WN, 0);
    gemm_bf16_kernel<<<ggrid(B_ * WN), TB, 0, stream>>>(XQP, WP21, s2b1, H, B_ * WN, 1);
    pack_rows_kernel<<<blk(B_ * WN * (Q_ / 2), TB), TB, 0, stream>>>(H, HP, B_ * WN * (Q_ / 2));
    gemm_bf16_kernel<<<ggrid(B_ * WN), TB, 0, stream>>>(HP, WP22, s2b2, ENC2, B_ * WN, 0);

    // sim (f32 WMMA), batched over grid.z
    {
        dim3 g(blk((T_ / 16) * (WN / 16), TB / 32), 1, B_);
        sim_wmma_kernel<<<g, TB, 0, stream>>>(V, ENC1, SIM);
    }
    // sliding sums (cs = 8, 16) for v and sim, then labels
    sliding_sum_kernel<<<blk(B_ * C_, TB), TB, 0, stream>>>(V, SS8, C_, 8);
    sliding_sum_kernel<<<blk(B_ * C_, TB), TB, 0, stream>>>(V, SS16, C_, 16);
    sliding_sum_kernel<<<blk(B_ * WN, TB), TB, 0, stream>>>(SIM, SM8, WN, 8);
    sliding_sum_kernel<<<blk(B_ * WN, TB), TB, 0, stream>>>(SIM, SM16, WN, 16);
    argmax_kernel<<<blk(B_ * T_, TB), TB, 0, stream>>>(SM8, LAB8, 8);
    argmax_kernel<<<blk(B_ * T_, TB), TB, 0, stream>>>(SM16, LAB16, 16);

    // jj = 0 : ws=64, cs=8
    {
        dim3 g(ggrid(ROWS0), 1, B_);
        proj_gemm_kernel<<<g, TB, 0, stream>>>(SS8, LAB8, ENC2, WPpj, pb, MBUF, ROWS0, 8);
        pool_kernel<<<dim3(P_, B_), TB, 0, stream>>>(MBUF, vlen, POOL, ROWS0);
        seg_head_kernel<<<blk(B_ * 2 * P_, TB), TB, 0, stream>>>(POOL, sgw0, sgb0, out + 0);
        vec_head_kernel<<<dim3(P_, B_), TB, 0, stream>>>(POOL, stw0, stb0, out + 512);
        vec_head_kernel<<<dim3(P_, B_), TB, 0, stream>>>(POOL, enw0, enb0, out + 768);
    }
    // jj = 1 : ws=128, cs=16
    {
        dim3 g(ggrid(ROWS1), 1, B_);
        proj_gemm_kernel<<<g, TB, 0, stream>>>(SS16, LAB16, ENC2, WPpj, pb, MBUF, ROWS1, 16);
        pool_kernel<<<dim3(P_, B_), TB, 0, stream>>>(MBUF, vlen, POOL, ROWS1);
        seg_head_kernel<<<blk(B_ * 2 * P_, TB), TB, 0, stream>>>(POOL, sgw1, sgb1, out + 1024);
        vec_head_kernel<<<dim3(P_, B_), TB, 0, stream>>>(POOL, stw1, stb1, out + 1536);
        vec_head_kernel<<<dim3(P_, B_), TB, 0, stream>>>(POOL, enw1, enb1, out + 1792);
    }
}
